// TransformerPatchGenerator_86560770883962
// MI455X (gfx1250) — matrix operand order repaired and checked by reading, not compiled, not run on hardware
//
#include <hip/hip_runtime.h>
#include <hip/hip_bf16.h>
#include <math.h>

// ---------------------------------------------------------------------------
// Model constants (fixed by the reference)
// ---------------------------------------------------------------------------
#define CB    128      // batch
#define NPG   128      // nodes per group
#define CF    256      // input features
#define CD    192      // model dim
#define CNH   8        // heads
#define CDH   24       // head dim
#define CL    4        // layers
#define CDFF  512
#define CP    32       // patch steps
#define CEH   96       // gat1 per-head channels
#define CGH   4        // gat1 heads
#define CN    (CB*NPG) // 16384 nodes

typedef __attribute__((ext_vector_type(16))) _Float16 v16h;
typedef __attribute__((ext_vector_type(8)))  float    v8f;

// ---------------------------------------------------------------------------
// WMMA GEMM:  C[M,N] = act( A[M,K] @ Wh[N,K]^T + bias )
//   A fp32 (converted to f16 while staging), Wh pre-converted f16 weights
//   staged into LDS with global_load_async_to_lds_b128 (ASYNCcnt protocol).
//   M%64==0, N%64==0, K%32==0 guaranteed by caller.
//   ACT: 0 = none, 1 = exact GELU.  Rows padded to 40 halves (80B) so the
//   16 fragment rows read by lanes 0..15 land in distinct LDS banks.
//   All staging loops have static trip counts (128 threads): no per-lane
//   predication / exec-mask traffic in the K-loop.
// ---------------------------------------------------------------------------
#define TM 64
#define TN 64
#define TK 32
#define LPAD 40

template <bool HASBIAS, int ACT>
__global__ __launch_bounds__(128) void gemm_f16wmma(
    const float* __restrict__ A, const _Float16* __restrict__ Wh,
    const float* __restrict__ bias, float* __restrict__ C,
    int M, int N, int K, int ldc)
{
  __shared__ _Float16 sA[TM][LPAD];
  __shared__ _Float16 sB[TN][LPAD];

  const int tid  = threadIdx.x;
  const int wave = tid >> 5;
  const int lane = tid & 31;
  const int m0   = blockIdx.x * TM;
  const int n0   = blockIdx.y * TN;

  v8f acc[4];
#pragma unroll
  for (int i = 0; i < 4; ++i) acc[i] = (v8f)(0.0f);

  for (int k0 = 0; k0 < K; k0 += TK) {
    // --- B tile: async DMA of pre-converted f16 weights into LDS.
    // 64 rows x 32 halves = 256 x 16B chunks; exactly 2 per thread.
#pragma unroll
    for (int it = 0; it < 2; ++it) {
      const int q = tid + it * 128;               // 0..255, always in range
      const int r = q >> 2, part = q & 3;         // part: 16B chunk in row
      unsigned lds = (unsigned)(size_t)&sB[r][part * 8];
      unsigned long long ga =
          (unsigned long long)(size_t)(Wh + (size_t)(n0 + r) * K + k0 + part * 8);
      asm volatile("global_load_async_to_lds_b128 %0, %1, off"
                   :: "v"(lds), "v"(ga) : "memory");
    }

    // --- A tile: vector-load fp32, convert to f16 in LDS (overlaps the DMA).
    // 64 rows x 32 floats = 512 float4 chunks; exactly 4 per thread.
#pragma unroll
    for (int it = 0; it < 4; ++it) {
      const int i = tid + it * 128;               // 0..511, always in range
      const int r = i >> 3, c4 = (i & 7) << 2;
      const float4 v = *(const float4*)&A[(size_t)(m0 + r) * K + (k0 + c4)];
      sA[r][c4 + 0] = (_Float16)v.x;
      sA[r][c4 + 1] = (_Float16)v.y;
      sA[r][c4 + 2] = (_Float16)v.z;
      sA[r][c4 + 3] = (_Float16)v.w;
      if (k0 + TK < K)
        __builtin_prefetch(&A[(size_t)(m0 + r) * K + (k0 + TK + c4)], 0, 0);
    }

    // Each wave drains its own async copies, then the workgroup barrier
    // makes every wave's LDS writes visible to all.
    asm volatile("s_wait_asynccnt 0" ::: "memory");
    __syncthreads();

    // A fragment (ISA 7.12.2, 16-bit A 16x32): lanes 0-15 row M=lane,
    // halves 0..7 = K 0..7, 8..15 = K 16..23; lanes 16-31 K offset +8/+24.
    v16h af;
    {
      int row = (wave << 4) + (lane & 15);
      int kh  = (lane >> 4) << 3;   // 0 or 8
#pragma unroll
      for (int t = 0; t < 8; ++t) {
        af[t]     = sA[row][kh + t];
        af[t + 8] = sA[row][kh + 16 + t];
      }
    }
#pragma unroll
    for (int nt = 0; nt < 4; ++nt) {
      // B fragment (K-major 32x16): lanes 0-15 col N=lane K 0..7,16..23,
      // lanes 16-31 col N=lane-16 K 8..15,24..31. B[k][n] = W[n][k].
      v16h bf;
      int col = (nt << 4) + (lane & 15);
      int kb  = (lane >> 4) << 3;   // 0 or 8
#pragma unroll
      for (int t = 0; t < 8; ++t) {
        bf[t]     = sB[col][kb + t];
        bf[t + 8] = sB[col][kb + 16 + t];
      }

      acc[nt] = __builtin_amdgcn_wmma_f32_16x16x32_f16(
          false, af, false, bf, (short)0, acc[nt], false, false);
    }
    __syncthreads();
  }

  // C/D layout: VGPR j -> row = j + 8*lane[4], col = lane&15.
#pragma unroll
  for (int nt = 0; nt < 4; ++nt) {
    int col = n0 + (nt << 4) + (lane & 15);
    float bv = HASBIAS ? bias[col] : 0.0f;
#pragma unroll
    for (int j = 0; j < 8; ++j) {
      int row = m0 + (wave << 4) + j + ((lane >> 4) << 3);
      float v = acc[nt][j] + bv;
      if (ACT == 1) v = 0.5f * v * (1.0f + erff(v * 0.70710678118654752f));
      C[(size_t)row * ldc + col] = v;
    }
  }
}

// ---------------------------------------------------------------------------
// Small helper kernels
// ---------------------------------------------------------------------------
__global__ void fill_kernel(float* p, float v, long n) {
  long i = (long)blockIdx.x * blockDim.x + threadIdx.x;
  for (; i < n; i += (long)gridDim.x * blockDim.x) p[i] = v;
}

__global__ void f32to16_kernel(const float* __restrict__ in,
                               _Float16* __restrict__ out, long n) {
  long i = (long)blockIdx.x * blockDim.x + threadIdx.x;
  for (; i < n; i += (long)gridDim.x * blockDim.x) out[i] = (_Float16)in[i];
}

__device__ __forceinline__ void atomicMaxF(float* a, float v) {
  int old = __float_as_int(*a);
  while (__int_as_float(old) < v) {
    int assumed = old;
    old = atomicCAS((int*)a, assumed, __float_as_int(v));
    if (old == assumed) break;
  }
}

// per-node attention logits: als[n,h] = <xp[n,h,:], att_src[h]>, same for dst
__global__ void att_logits_kernel(const float* __restrict__ xp,
                                  const float* __restrict__ as,
                                  const float* __restrict__ ad,
                                  float* __restrict__ als, float* __restrict__ ald,
                                  int Nn, int heads, int ch)
{
  int id = blockIdx.x * blockDim.x + threadIdx.x;
  if (id >= Nn * heads) return;
  int n = id / heads, h = id % heads;
  const float* xr = xp + (size_t)n * heads * ch + h * ch;
  const float* s  = as + h * ch;
  const float* d  = ad + h * ch;
  float vs = 0.f, vd = 0.f;
  for (int c = 0; c < ch; ++c) { vs += xr[c] * s[c]; vd += xr[c] * d[c]; }
  als[id] = vs; ald[id] = vd;
}

// edge pass 1: e = leaky_relu(als[s]+ald[d]); segment max over dst
__global__ void edge_max_kernel(const float* als, const float* ald,
                                const int* src, const int* dst,
                                float* e, float* emax, int E, int Nn, int heads)
{
  long id = (long)blockIdx.x * blockDim.x + threadIdx.x;
  long total = (long)(E + Nn) * heads;
  if (id >= total) return;
  int j = (int)(id / heads), h = (int)(id % heads);
  int sN = (j < E) ? src[j] : (j - E);
  int dN = (j < E) ? dst[j] : (j - E);
  float v = als[sN * heads + h] + ald[dN * heads + h];
  v = (v > 0.f) ? v : 0.2f * v;
  e[id] = v;
  atomicMaxF(&emax[dN * heads + h], v);
}

// edge pass 2: ex = exp(e - emax[d]); segment sum
__global__ void edge_exp_kernel(float* e, const float* emax,
                                const int* src, const int* dst,
                                float* den, int E, int Nn, int heads)
{
  long id = (long)blockIdx.x * blockDim.x + threadIdx.x;
  long total = (long)(E + Nn) * heads;
  if (id >= total) return;
  int j = (int)(id / heads), h = (int)(id % heads);
  int dN = (j < E) ? dst[j] : (j - E);
  float ex = expf(e[id] - emax[dN * heads + h]);
  e[id] = ex;
  atomicAdd(&den[dN * heads + h], ex);
}

// edge pass 3: out[d] += xp[s] * coef ; one block per edge, thread per (h,c)
__global__ void edge_aggr_kernel(const float* __restrict__ xp,
                                 const float* __restrict__ ex,
                                 const float* __restrict__ den,
                                 const int* src, const int* dst,
                                 float* out, int E, int Nn, int heads, int ch)
{
  int j = blockIdx.x;
  int h = threadIdx.x / ch, c = threadIdx.x % ch;
  int sN = (j < E) ? src[j] : (j - E);
  int dN = (j < E) ? dst[j] : (j - E);
  float coef = ex[(size_t)j * heads + h] / (den[dN * heads + h] + 1e-16f);
  atomicAdd(&out[(size_t)dN * heads * ch + h * ch + c],
            xp[(size_t)sN * heads * ch + h * ch + c] * coef);
}

// out = act(in + bias[col])  (act 0=none, 2=ELU)
__global__ void bias_act_kernel(const float* in, const float* bias, float* out,
                                long total, int C, int act)
{
  long i = (long)blockIdx.x * blockDim.x + threadIdx.x;
  if (i >= total) return;
  float v = in[i] + bias[i % C];
  if (act == 2) v = (v > 0.f) ? v : expm1f(v);
  out[i] = v;
}

// sinusoidal positional encoding pe[(P+1) x D]
__global__ void pe_kernel(float* pe) {
  int id = blockIdx.x * blockDim.x + threadIdx.x;
  if (id >= (CP + 1) * CD) return;
  int t = id / CD, d = id % CD;
  int i2 = d & ~1;
  float div = expf(-(float)i2 * (logf(10000.f) / (float)CD));
  float ang = (float)t * div;
  pe[id] = (d & 1) ? cosf(ang) : sinf(ang);
}

// tgt[b,0,:] = mem[b*NPG + tli[b], :]
__global__ void init_tgt_kernel(const float* mem, const int* tli, float* tgt) {
  int id = blockIdx.x * blockDim.x + threadIdx.x;
  if (id >= CB * CD) return;
  int b = id / CD, d = id % CD;
  tgt[((size_t)b * (CP + 1)) * CD + d] =
      mem[((size_t)b * NPG + tli[b]) * CD + d];
}

// x[b*s+t, :] = tgt[b,t,:] + pe[t,:]
__global__ void build_x_kernel(const float* tgt, const float* pe, float* x, int s) {
  long id = (long)blockIdx.x * blockDim.x + threadIdx.x;
  long total = (long)CB * s * CD;
  if (id >= total) return;
  long r = id / CD; int d = (int)(id % CD);
  int b = (int)(r / s), t = (int)(r % s);
  x[id] = tgt[((size_t)b * (CP + 1) + t) * CD + d] + pe[t * CD + d];
}

// causal self-attention, qkv packed (q|k|v) per row, thread per (b,h,t)
__global__ void self_attn_kernel(const float* __restrict__ qkv,
                                 float* __restrict__ out, int s)
{
  int id = blockIdx.x * blockDim.x + threadIdx.x;
  if (id >= CB * CNH * s) return;
  int b = id / (CNH * s); int rem = id % (CNH * s);
  int h = rem / s, t = rem % s;
  const float* q = qkv + ((size_t)(b * s + t) * (3 * CD)) + h * CDH;
  const float scale = 0.20412414523193154f; // 1/sqrt(24)
  float sc[CP]; float mx = -1e30f;
  for (int k = 0; k <= t; ++k) {
    const float* kp = qkv + ((size_t)(b * s + k) * (3 * CD)) + CD + h * CDH;
    float dq = 0.f;
    for (int c = 0; c < CDH; ++c) dq += q[c] * kp[c];
    dq *= scale; sc[k] = dq; mx = fmaxf(mx, dq);
  }
  float den = 0.f;
  for (int k = 0; k <= t; ++k) { sc[k] = expf(sc[k] - mx); den += sc[k]; }
  float inv = 1.f / den;
  float a[CDH];
  for (int c = 0; c < CDH; ++c) a[c] = 0.f;
  for (int k = 0; k <= t; ++k) {
    const float* vp = qkv + ((size_t)(b * s + k) * (3 * CD)) + 2 * CD + h * CDH;
    float w = sc[k] * inv;
    for (int c = 0; c < CDH; ++c) a[c] += w * vp[c];
  }
  float* o = out + ((size_t)(b * s + t) * CD) + h * CDH;
  for (int c = 0; c < CDH; ++c) o[c] = a[c];
}

// cross-attention over 128 precomputed memory K/V slots (kv: 16384 x 384)
__global__ void cross_attn_kernel(const float* __restrict__ q,
                                  const float* __restrict__ kv,
                                  float* __restrict__ out, int s)
{
  int id = blockIdx.x * blockDim.x + threadIdx.x;
  if (id >= CB * CNH * s) return;
  int b = id / (CNH * s); int rem = id % (CNH * s);
  int h = rem / s, t = rem % s;
  const float* qp = q + ((size_t)(b * s + t) * CD) + h * CDH;
  const float scale = 0.20412414523193154f;
  float mx = -1e30f;
  for (int j = 0; j < NPG; ++j) {
    const float* kp = kv + ((size_t)(b * NPG + j) * (2 * CD)) + h * CDH;
    float d = 0.f;
    for (int c = 0; c < CDH; ++c) d += qp[c] * kp[c];
    mx = fmaxf(mx, d * scale);
  }
  float den = 0.f, a[CDH];
  for (int c = 0; c < CDH; ++c) a[c] = 0.f;
  for (int j = 0; j < NPG; ++j) {
    const float* kp = kv + ((size_t)(b * NPG + j) * (2 * CD)) + h * CDH;
    float d = 0.f;
    for (int c = 0; c < CDH; ++c) d += qp[c] * kp[c];
    float w = expf(d * scale - mx); den += w;
    const float* vp = kp + CD;
    for (int c = 0; c < CDH; ++c) a[c] += w * vp[c];
  }
  float inv = 1.f / den;
  float* o = out + ((size_t)(b * s + t) * CD) + h * CDH;
  for (int c = 0; c < CDH; ++c) o[c] = a[c] * inv;
}

// x = LayerNorm(x + sub) * g + b   (one block of 192 threads per row)
__global__ void ln_kernel(float* x, const float* sub,
                          const float* g, const float* b, int R)
{
  int row = blockIdx.x; int t = threadIdx.x;
  __shared__ float red[CD];
  float v = x[(size_t)row * CD + t] + sub[(size_t)row * CD + t];
  red[t] = v; __syncthreads();
  for (int off = 128; off > 0; off >>= 1) {
    if (t < off && t + off < CD) red[t] += red[t + off];
    __syncthreads();
  }
  float mean = red[0] / (float)CD; __syncthreads();
  float dv = v - mean;
  red[t] = dv * dv; __syncthreads();
  for (int off = 128; off > 0; off >>= 1) {
    if (t < off && t + off < CD) red[t] += red[t + off];
    __syncthreads();
  }
  float var = red[0] / (float)CD;
  x[(size_t)row * CD + t] = dv * rsqrtf(var + 1e-5f) * g[t] + b[t];
}

// last[b,:] = x[b,s-1,:];  tgt[b,s,:] = last
__global__ void tail_update_kernel(const float* x, float* last, float* tgt, int s) {
  int id = blockIdx.x * blockDim.x + threadIdx.x;
  if (id >= CB * CD) return;
  int b = id / CD, d = id % CD;
  float v = x[((size_t)b * s + (s - 1)) * CD + d];
  last[id] = v;
  tgt[((size_t)b * (CP + 1) + s) * CD + d] = v;
}

// cl[b,t] = <x[b,t,:], last[b,:]> scattered into adj workspace (reference at[])
__global__ void cl_adj_kernel(const float* x, const float* last,
                              float* adjws, int s, int i)
{
  int id = blockIdx.x * blockDim.x + threadIdx.x;
  if (id >= CB * s) return;
  int b = id / s, t = id % s;
  const float* xp = x + ((size_t)b * s + t) * CD;
  const float* lp = last + (size_t)b * CD;
  float d = 0.f;
  for (int c = 0; c < CD; ++c) d += xp[c] * lp[c];
  float* A = adjws + (size_t)b * (CP + 1) * (CP + 1);
  if (t == 0) { A[i * (CP + 1) + CP] = d; A[CP * (CP + 1) + i] = d; }
  if (i > 0 && t >= 1 && t <= i) {
    A[i * (CP + 1) + (t - 1)] = d;
    A[(t - 1) * (CP + 1) + i] = d;
  }
}

// adj = triu(ws,1) + triu(ws,1)^T
__global__ void adj_final_kernel(const float* ws, float* out) {
  int id = blockIdx.x * blockDim.x + threadIdx.x;
  const int S = (CP + 1) * (CP + 1);
  if (id >= CB * S) return;
  int b = id / S; int rc = id % S;
  int r = rc / (CP + 1), c = rc % (CP + 1);
  const float* A = ws + (size_t)b * S;
  float v = (r < c) ? A[r * (CP + 1) + c] : 0.f;
  v += (c < r) ? A[c * (CP + 1) + r] : 0.f;
  out[id] = v;
}

// ---------------------------------------------------------------------------
// Host orchestration
// ---------------------------------------------------------------------------
extern "C" void kernel_launch(void* const* d_in, const int* in_sizes, int n_in,
                              void* d_out, int out_size, void* d_ws, size_t ws_size,
                              hipStream_t stream)
{
  const float* x0       = (const float*)d_in[0];
  const int*   eidx     = (const int*)d_in[1];
  const int*   tli      = (const int*)d_in[2];
  const float* gat1_w   = (const float*)d_in[5];
  const float* gat1_as  = (const float*)d_in[6];
  const float* gat1_ad  = (const float*)d_in[7];
  const float* gat1_b   = (const float*)d_in[8];
  const float* gat2_w   = (const float*)d_in[9];
  const float* gat2_as  = (const float*)d_in[10];
  const float* gat2_ad  = (const float*)d_in[11];
  const float* gat2_b   = (const float*)d_in[12];
  const float* sa_qkv_w = (const float*)d_in[13];
  const float* sa_qkv_b = (const float*)d_in[14];
  const float* sa_out_w = (const float*)d_in[15];
  const float* sa_out_b = (const float*)d_in[16];
  const float* ca_qkv_w = (const float*)d_in[17];
  const float* ca_qkv_b = (const float*)d_in[18];
  const float* ca_out_w = (const float*)d_in[19];
  const float* ca_out_b = (const float*)d_in[20];
  const float* ff1_w    = (const float*)d_in[21];
  const float* ff1_b    = (const float*)d_in[22];
  const float* ff2_w    = (const float*)d_in[23];
  const float* ff2_b    = (const float*)d_in[24];
  const float* ln_g     = (const float*)d_in[25];
  const float* ln_b     = (const float*)d_in[26];
  const float* h1_w     = (const float*)d_in[27];
  const float* h1_b     = (const float*)d_in[28];
  const float* h2_w     = (const float*)d_in[29];
  const float* h2_b     = (const float*)d_in[30];

  const int E = in_sizes[1] / 2;           // 262144
  const int* src = eidx;
  const int* dst = eidx + E;

  // --- workspace carve-out ---
  char* cur = (char*)d_ws;
  auto alloc = [&](size_t bytes) {
    void* p = (void*)cur;
    cur += (bytes + 255) & ~(size_t)255;
    return p;
  };
  float* xp1   = (float*)alloc((size_t)CN * 384 * 4);
  float* agg1  = (float*)alloc((size_t)CN * 384 * 4);
  float* hbuf  = (float*)alloc((size_t)CN * 384 * 4);
  float* als   = (float*)alloc((size_t)CN * CGH * 4);
  float* ald   = (float*)alloc((size_t)CN * CGH * 4);
  float* ebuf  = (float*)alloc((size_t)(E + CN) * CGH * 4);
  float* emax  = (float*)alloc((size_t)CN * CGH * 4);
  float* den   = (float*)alloc((size_t)CN * CGH * 4);
  float* xp2   = (float*)alloc((size_t)CN * CD * 4);
  float* agg2  = (float*)alloc((size_t)CN * CD * 4);
  float* mem   = (float*)alloc((size_t)CN * CD * 4);
  float* ca_kv = (float*)alloc((size_t)CL * CN * 2 * CD * 4);
  float* pe    = (float*)alloc((size_t)(CP + 1) * CD * 4);
  float* tgt   = (float*)alloc((size_t)CB * (CP + 1) * CD * 4);
  float* xbuf  = (float*)alloc((size_t)CB * CP * CD * 4);
  float* qkvb  = (float*)alloc((size_t)CB * CP * 3 * CD * 4);
  float* attnb = (float*)alloc((size_t)CB * CP * CD * 4);
  float* subb  = (float*)alloc((size_t)CB * CP * CD * 4);
  float* qbuf  = (float*)alloc((size_t)CB * CP * CD * 4);
  float* ffb   = (float*)alloc((size_t)CB * CP * CDFF * 4);
  float* lastb = (float*)alloc((size_t)CB * CD * 4);
  float* hfeat = (float*)alloc((size_t)CB * CD * 4);
  float* adjws = (float*)alloc((size_t)CB * (CP + 1) * (CP + 1) * 4);

  auto cvt16 = [&](const float* srcp, long n) -> _Float16* {
    _Float16* p = (_Float16*)alloc((size_t)n * sizeof(_Float16));
    int blocks = (int)((n + 255) / 256); if (blocks > 8192) blocks = 8192;
    f32to16_kernel<<<blocks, 256, 0, stream>>>(srcp, p, n);
    return p;
  };

  // Pre-convert all GEMM weights to f16 once per launch (reused 32x-128x).
  _Float16* w_gat1 = cvt16(gat1_w, (long)(CGH * CEH) * CF);
  _Float16* w_gat2 = cvt16(gat2_w, (long)CD * (CGH * CEH));
  _Float16* w_saq  = cvt16(sa_qkv_w, (long)CL * 3 * CD * CD);
  _Float16* w_sao  = cvt16(sa_out_w, (long)CL * CD * CD);
  _Float16* w_caq  = cvt16(ca_qkv_w, (long)CL * 3 * CD * CD);
  _Float16* w_cao  = cvt16(ca_out_w, (long)CL * CD * CD);
  _Float16* w_ff1  = cvt16(ff1_w, (long)CL * CDFF * CD);
  _Float16* w_ff2  = cvt16(ff2_w, (long)CL * CD * CDFF);
  _Float16* w_h1   = cvt16(h1_w, (long)CD * CD);
  _Float16* w_h2   = cvt16(h2_w, (long)CF * CD);

  auto gemm = [&](const float* A, const _Float16* Wh, const float* bias, float* C,
                  int M, int Nc, int K, int ldc, int act) {
    dim3 grid(M / TM, Nc / TN);
    if (bias) {
      if (act == 1)
        gemm_f16wmma<true, 1><<<grid, 128, 0, stream>>>(A, Wh, bias, C, M, Nc, K, ldc);
      else
        gemm_f16wmma<true, 0><<<grid, 128, 0, stream>>>(A, Wh, bias, C, M, Nc, K, ldc);
    } else {
      gemm_f16wmma<false, 0><<<grid, 128, 0, stream>>>(A, Wh, bias, C, M, Nc, K, ldc);
    }
  };
  auto fill = [&](float* p, float v, long n) {
    int blocks = (int)((n + 255) / 256); if (blocks > 4096) blocks = 4096;
    fill_kernel<<<blocks, 256, 0, stream>>>(p, v, n);
  };

  // ===================== GAT layer 1 (4 heads x 96) =====================
  gemm(x0, w_gat1, nullptr, xp1, CN, CGH * CEH, CF, CGH * CEH, 0);
  att_logits_kernel<<<(CN * CGH + 255) / 256, 256, 0, stream>>>(
      xp1, gat1_as, gat1_ad, als, ald, CN, CGH, CEH);
  fill(emax, -3.0e38f, (long)CN * CGH);
  fill(den, 0.f, (long)CN * CGH);
  fill(agg1, 0.f, (long)CN * 384);
  {
    long tot = (long)(E + CN) * CGH;
    edge_max_kernel<<<(int)((tot + 255) / 256), 256, 0, stream>>>(
        als, ald, src, dst, ebuf, emax, E, CN, CGH);
    edge_exp_kernel<<<(int)((tot + 255) / 256), 256, 0, stream>>>(
        ebuf, emax, src, dst, den, E, CN, CGH);
    edge_aggr_kernel<<<E + CN, CGH * CEH, 0, stream>>>(
        xp1, ebuf, den, src, dst, agg1, E, CN, CGH, CEH);
  }
  bias_act_kernel<<<(int)(((long)CN * 384 + 255) / 256), 256, 0, stream>>>(
      agg1, gat1_b, hbuf, (long)CN * 384, 384, 2 /*ELU*/);

  // ===================== GAT layer 2 (1 head x 192) =====================
  gemm(hbuf, w_gat2, nullptr, xp2, CN, CD, CGH * CEH, CD, 0);
  att_logits_kernel<<<(CN + 255) / 256, 256, 0, stream>>>(
      xp2, gat2_as, gat2_ad, als, ald, CN, 1, CD);
  fill(emax, -3.0e38f, (long)CN);
  fill(den, 0.f, (long)CN);
  fill(agg2, 0.f, (long)CN * CD);
  {
    long tot = (long)(E + CN);
    edge_max_kernel<<<(int)((tot + 255) / 256), 256, 0, stream>>>(
        als, ald, src, dst, ebuf, emax, E, CN, 1);
    edge_exp_kernel<<<(int)((tot + 255) / 256), 256, 0, stream>>>(
        ebuf, emax, src, dst, den, E, CN, 1);
    edge_aggr_kernel<<<E + CN, CD, 0, stream>>>(
        xp2, ebuf, den, src, dst, agg2, E, CN, 1, CD);
  }
  bias_act_kernel<<<(int)(((long)CN * CD + 255) / 256), 256, 0, stream>>>(
      agg2, gat2_b, mem, (long)CN * CD, CD, 0);

  // ====== hoisted cross-attention K/V (loop-invariant: 32x GEMM saving) ======
  for (int l = 0; l < CL; ++l)
    gemm(mem, w_caq + (size_t)l * 3 * CD * CD + (size_t)CD * CD,
         ca_qkv_b + (size_t)l * 3 * CD + CD,
         ca_kv + (size_t)l * CN * 2 * CD, CN, 2 * CD, CD, 2 * CD, 0);

  pe_kernel<<<((CP + 1) * CD + 255) / 256, 256, 0, stream>>>(pe);
  init_tgt_kernel<<<(CB * CD + 255) / 256, 256, 0, stream>>>(mem, tli, tgt);
  fill(adjws, 0.f, (long)CB * (CP + 1) * (CP + 1));

  float* feats_out = (float*)d_out;                        // (B, P, F)
  float* adj_out   = (float*)d_out + (size_t)CB * CP * CF; // (B, P+1, P+1)

  // ===================== autoregressive decoder =====================
  for (int i = 0; i < CP; ++i) {
    const int s = i + 1;
    const int R = CB * s;                 // rows, multiple of 128
    build_x_kernel<<<(int)(((long)R * CD + 255) / 256), 256, 0, stream>>>(
        tgt, pe, xbuf, s);

    for (int l = 0; l < CL; ++l) {
      const float* g0 = ln_g + ((size_t)l * 3 + 0) * CD;
      const float* b0 = ln_b + ((size_t)l * 3 + 0) * CD;
      const float* g1 = ln_g + ((size_t)l * 3 + 1) * CD;
      const float* b1 = ln_b + ((size_t)l * 3 + 1) * CD;
      const float* g2 = ln_g + ((size_t)l * 3 + 2) * CD;
      const float* b2 = ln_b + ((size_t)l * 3 + 2) * CD;

      // --- causal self-attention ---
      gemm(xbuf, w_saq + (size_t)l * 3 * CD * CD, sa_qkv_b + (size_t)l * 3 * CD,
           qkvb, R, 3 * CD, CD, 3 * CD, 0);
      self_attn_kernel<<<(CB * CNH * s + 127) / 128, 128, 0, stream>>>(qkvb, attnb, s);
      gemm(attnb, w_sao + (size_t)l * CD * CD, sa_out_b + (size_t)l * CD,
           subb, R, CD, CD, CD, 0);
      ln_kernel<<<R, CD, 0, stream>>>(xbuf, subb, g0, b0, R);

      // --- cross-attention (K/V hoisted) ---
      gemm(xbuf, w_caq + (size_t)l * 3 * CD * CD, ca_qkv_b + (size_t)l * 3 * CD,
           qbuf, R, CD, CD, CD, 0);
      cross_attn_kernel<<<(CB * CNH * s + 127) / 128, 128, 0, stream>>>(
          qbuf, ca_kv + (size_t)l * CN * 2 * CD, attnb, s);
      gemm(attnb, w_cao + (size_t)l * CD * CD, ca_out_b + (size_t)l * CD,
           subb, R, CD, CD, CD, 0);
      ln_kernel<<<R, CD, 0, stream>>>(xbuf, subb, g1, b1, R);

      // --- FFN (GELU fused into GEMM epilogue) ---
      gemm(xbuf, w_ff1 + (size_t)l * CDFF * CD, ff1_b + (size_t)l * CDFF,
           ffb, R, CDFF, CD, CDFF, 1);
      gemm(ffb, w_ff2 + (size_t)l * CD * CDFF, ff2_b + (size_t)l * CD,
           subb, R, CD, CDFF, CD, 0);
      ln_kernel<<<R, CD, 0, stream>>>(xbuf, subb, g2, b2, R);
    }

    // last token -> feats head + append to tgt
    tail_update_kernel<<<(CB * CD + 255) / 256, 256, 0, stream>>>(xbuf, lastb, tgt, s);
    gemm(lastb, w_h1, h1_b, hfeat, CB, CD, CD, CD, 1);
    gemm(hfeat, w_h2, h2_b, feats_out + (size_t)i * CF, CB, CF, CD, CP * CF, 0);
    cl_adj_kernel<<<(CB * s + 127) / 128, 128, 0, stream>>>(xbuf, lastb, adjws, s, i);
  }

  adj_final_kernel<<<(CB * (CP + 1) * (CP + 1) + 255) / 256, 256, 0, stream>>>(
      adjws, adj_out);
}
